// Head_51427938402332
// MI455X (gfx1250) — matrix operand order repaired and checked by reading, not compiled, not run on hardware
//
#include <hip/hip_runtime.h>
#include <hip/hip_bf16.h>

typedef _Float16 h16;
typedef __attribute__((ext_vector_type(16))) _Float16 v16h;
typedef __attribute__((ext_vector_type(8)))  _Float16 v8h;
typedef __attribute__((ext_vector_type(8)))  float    v8f;

#define SEQ    4096
#define DMODEL 1024
#define HEAD   128
#define NBATCH 4
#define ATTN_WAVES 4

// ---------------------------------------------------------------------------
// Kernel 0: convert Wq/Wk/Wv (fp32 [1024][128]) -> fp16 transposed [3][128][1024]
// so the GEMM B-fragments become 32B-contiguous loads.
// ---------------------------------------------------------------------------
__global__ __launch_bounds__(256) void wt_kernel(
    const float* __restrict__ Wq, const float* __restrict__ Wk,
    const float* __restrict__ Wv, h16* __restrict__ Wt) {
  const int which = blockIdx.y;
  const float* W = which == 0 ? Wq : (which == 1 ? Wk : Wv);
  const int idx = blockIdx.x * 256 + threadIdx.x;   // 0..131071
  const int k = idx >> 7;                           // 0..1023
  const int n = idx & 127;                          // 0..127
  Wt[(size_t)which * HEAD * DMODEL + (size_t)n * DMODEL + k] =
      (h16)W[(size_t)k * HEAD + n];
}

// ---------------------------------------------------------------------------
// Kernel 1: QKV projection. One block = one 16-row tile of x (staged in LDS
// as fp16). 8 waves cover the 128 output columns; each wave keeps 3
// accumulators (Q,K,V) and runs v_wmma_f32_16x16x32_f16 over K=1024.
// Q,K stored row-major fp16 [B*T,128]; V stored transposed [B][128][T].
// ---------------------------------------------------------------------------
__global__ __launch_bounds__(256) void qkv_kernel(
    const float* __restrict__ x, const h16* __restrict__ Wt,
    h16* __restrict__ Qo, h16* __restrict__ Ko, h16* __restrict__ Vt) {
  __shared__ __align__(32) h16 xl[16 * DMODEL];     // 32 KB
  const int tid  = threadIdx.x;
  const int lane = tid & 31;
  const int wave = tid >> 5;
  const int mbase = blockIdx.x * 16;

  // cooperative x tile -> LDS (fp32 -> fp16), fully coalesced float4 loads
  const float4* x4 = (const float4*)(x + (size_t)mbase * DMODEL);
#pragma unroll
  for (int i = 0; i < 16; ++i) {
    int idx = i * 256 + tid;                        // 0..4095 float4 slots
    float4 v = x4[idx];
    h16* dst = xl + idx * 4;
    dst[0] = (h16)v.x; dst[1] = (h16)v.y; dst[2] = (h16)v.z; dst[3] = (h16)v.w;
  }
  __syncthreads();

  const int half = lane >> 4;                       // K-half select (A/B layout)
  const int l15  = lane & 15;
  const int nbase = wave * 16;

  v8f acc0 = {}, acc1 = {}, acc2 = {};
  const h16* arow = xl + (size_t)l15 * DMODEL;                       // A row m = lane&15
  const h16* w0 = Wt + (size_t)(nbase + l15) * DMODEL + half * 16;   // B col n = lane&15
  const h16* w1 = w0 + (size_t)HEAD * DMODEL;
  const h16* w2 = w1 + (size_t)HEAD * DMODEL;

  for (int k0 = 0; k0 < DMODEL; k0 += 32) {
    // A fragment: elems 0..15 -> K = half*16 + e
    v16h a;
    v8h lo = *(const v8h*)(arow + k0 + half * 16);
    v8h hi = *(const v8h*)(arow + k0 + half * 16 + 8);
#pragma unroll
    for (int e = 0; e < 8; ++e) { a[e] = lo[e]; a[8 + e] = hi[e]; }
    // B fragments: elem e -> K = half*16 + e (contiguous 32B per lane)
    v16h b0 = *(const v16h*)(w0 + k0);
    v16h b1 = *(const v16h*)(w1 + k0);
    v16h b2 = *(const v16h*)(w2 + k0);
    acc0 = __builtin_amdgcn_wmma_f32_16x16x32_f16(false, a, false, b0, (short)0, acc0, false, false);
    acc1 = __builtin_amdgcn_wmma_f32_16x16x32_f16(false, a, false, b1, (short)0, acc1, false, false);
    acc2 = __builtin_amdgcn_wmma_f32_16x16x32_f16(false, a, false, b2, (short)0, acc2, false, false);
  }

  // C layout: elem i -> row m = i + 8*half, col n = lane&15
#pragma unroll
  for (int i = 0; i < 8; ++i) {
    const int m  = i + half * 8;
    const int gr = mbase + m;                       // global row in [0, B*T)
    Qo[(size_t)gr * HEAD + nbase + l15] = (h16)acc0[i];
    Ko[(size_t)gr * HEAD + nbase + l15] = (h16)acc1[i];
    const int b = gr >> 12;                         // / SEQ
    const int t = gr & (SEQ - 1);
    Vt[((size_t)b * HEAD + nbase + l15) * SEQ + t] = (h16)acc2[i];
  }
}

// ---------------------------------------------------------------------------
// Kernel 2: causal flash attention. One wave = one 16-query tile; loop over
// 32-key blocks with online softmax; fp16 WMMA, fp32 accumulation.
// ---------------------------------------------------------------------------
__global__ __launch_bounds__(32 * ATTN_WAVES) void attn_kernel(
    const h16* __restrict__ Q, const h16* __restrict__ K,
    const h16* __restrict__ Vt, float* __restrict__ out) {
  __shared__ __align__(32) h16 plds[ATTN_WAVES][16 * 32];  // P bounce buffers
  const int lane = threadIdx.x & 31;
  const int wave = threadIdx.x >> 5;
  const int tile = blockIdx.x * ATTN_WAVES + wave;  // 0 .. B*T/16-1
  const int b    = tile >> 8;                       // T/16 = 256 tiles/batch
  const int qt   = tile & 255;
  const int qbase = qt * 16;
  const int half  = lane >> 4;
  const int l15   = lane & 15;
  h16* pbuf = plds[wave];

  // Preload Q A-fragments for all 4 K-dim chunks (c0 = 0,32,64,96)
  const h16* qrow = Q + ((size_t)b * SEQ + qbase + l15) * HEAD;
  v16h qf[4];
#pragma unroll
  for (int c = 0; c < 4; ++c) {
    v8h lo = *(const v8h*)(qrow + c * 32 + half * 16);
    v8h hi = *(const v8h*)(qrow + c * 32 + half * 16 + 8);
#pragma unroll
    for (int e = 0; e < 8; ++e) { qf[c][e] = lo[e]; qf[c][8 + e] = hi[e]; }
  }

  v8f o[8];
#pragma unroll
  for (int i = 0; i < 8; ++i) o[i] = (v8f){};
  float mrun[8], lrun[8];
#pragma unroll
  for (int i = 0; i < 8; ++i) { mrun[i] = -1e30f; lrun[i] = 0.f; }

  const float scale = 0.0883883476483184f;          // 1/sqrt(128)
  const h16* kbat = K + (size_t)b * SEQ * HEAD;
  const h16* vbat = Vt + (size_t)b * HEAD * SEQ;

  const int nkb = (qbase + 16 + 31) >> 5;           // 32-key blocks to visit
  for (int kb = 0; kb < nkb; ++kb) {
    const int kbase = kb * 32;
    const bool do1 = (kbase + 16) <= (qbase + 15);

    // ---- S = Q K^T (two 16x16 tiles) ----
    v8f s0 = {}, s1 = {};
    {
      const h16* kr0 = kbat + (size_t)(kbase + l15) * HEAD + half * 16;
#pragma unroll
      for (int c = 0; c < 4; ++c) {
        v16h bf = *(const v16h*)(kr0 + c * 32);
        s0 = __builtin_amdgcn_wmma_f32_16x16x32_f16(false, qf[c], false, bf, (short)0, s0, false, false);
      }
    }
    if (do1) {
      const h16* kr1 = kbat + (size_t)(kbase + 16 + l15) * HEAD + half * 16;
#pragma unroll
      for (int c = 0; c < 4; ++c) {
        v16h bf = *(const v16h*)(kr1 + c * 32);
        s1 = __builtin_amdgcn_wmma_f32_16x16x32_f16(false, qf[c], false, bf, (short)0, s1, false, false);
      }
    }

    // ---- scale + causal mask (C layout: row = i + 8*half, col = lane&15) ----
#pragma unroll
    for (int i = 0; i < 8; ++i) {
      const int qg = qbase + i + half * 8;
      s0[i] = ((kbase + l15) <= qg) ? s0[i] * scale : -1e30f;
      s1[i] = (do1 && (kbase + 16 + l15) <= qg) ? s1[i] * scale : -1e30f;
    }

    // ---- online softmax: row max over 16-lane half-groups ----
    float rmax[8];
#pragma unroll
    for (int i = 0; i < 8; ++i) rmax[i] = fmaxf(s0[i], s1[i]);
#pragma unroll
    for (int d = 1; d < 16; d <<= 1)
#pragma unroll
      for (int i = 0; i < 8; ++i)
        rmax[i] = fmaxf(rmax[i], __shfl_xor(rmax[i], d, 32));

    float alpha[8];
#pragma unroll
    for (int i = 0; i < 8; ++i) {
      const float mn = fmaxf(mrun[i], rmax[i]);
      alpha[i] = __expf(mrun[i] - mn);
      mrun[i] = mn;
    }

    float rsum[8];
#pragma unroll
    for (int i = 0; i < 8; ++i) {
      s0[i] = __expf(s0[i] - mrun[i]);
      s1[i] = __expf(s1[i] - mrun[i]);
      rsum[i] = s0[i] + s1[i];
    }
#pragma unroll
    for (int d = 1; d < 16; d <<= 1)
#pragma unroll
      for (int i = 0; i < 8; ++i)
        rsum[i] += __shfl_xor(rsum[i], d, 32);

#pragma unroll
    for (int i = 0; i < 8; ++i) {
      lrun[i] = lrun[i] * alpha[i] + rsum[i];
#pragma unroll
      for (int ct = 0; ct < 8; ++ct) o[ct][i] *= alpha[i];
    }

    // ---- transpose P from C-layout to A-layout via wave-private LDS ----
#pragma unroll
    for (int i = 0; i < 8; ++i) {
      const int m = i + half * 8;
      pbuf[m * 32 + l15]      = (h16)s0[i];
      pbuf[m * 32 + 16 + l15] = (h16)s1[i];
    }
    asm volatile("s_wait_dscnt 0x0" ::: "memory");  // wave-local, no barrier
    v16h pf;
    {
      v8h lo = *(const v8h*)(pbuf + l15 * 32 + half * 16);
      v8h hi = *(const v8h*)(pbuf + l15 * 32 + half * 16 + 8);
#pragma unroll
      for (int e = 0; e < 8; ++e) { pf[e] = lo[e]; pf[8 + e] = hi[e]; }
    }

    // ---- O += P @ V (V^T rows give contiguous B-fragments) ----
#pragma unroll
    for (int ct = 0; ct < 8; ++ct) {
      const h16* vr = vbat + (size_t)(ct * 16 + l15) * SEQ + kbase + half * 16;
      v16h vf = *(const v16h*)vr;
      o[ct] = __builtin_amdgcn_wmma_f32_16x16x32_f16(false, pf, false, vf, (short)0, o[ct], false, false);
    }
  }

  // ---- epilogue: normalize and store fp32 ----
  float* orow = out + ((size_t)b * SEQ + qbase) * HEAD;
#pragma unroll
  for (int i = 0; i < 8; ++i) {
    const float inv = 1.0f / lrun[i];
    const int m = i + half * 8;
#pragma unroll
    for (int ct = 0; ct < 8; ++ct)
      orow[(size_t)m * HEAD + ct * 16 + l15] = o[ct][i] * inv;
  }
}

// ---------------------------------------------------------------------------
extern "C" void kernel_launch(void* const* d_in, const int* in_sizes, int n_in,
                              void* d_out, int out_size, void* d_ws, size_t ws_size,
                              hipStream_t stream) {
  const float* x  = (const float*)d_in[0];
  // d_in[1] = mask (unused; causality computed analytically)
  const float* Wq = (const float*)d_in[2];
  const float* Wk = (const float*)d_in[3];
  const float* Wv = (const float*)d_in[4];
  float* out = (float*)d_out;

  char* ws = (char*)d_ws;
  h16* Wt = (h16*)ws;                                   // 3*128*1024 fp16 = 768 KB
  h16* Qb = (h16*)(ws + 786432);                        // 4 MB
  h16* Kb = (h16*)(ws + 786432 + 4194304);              // 4 MB
  h16* Vt = (h16*)(ws + 786432 + 2 * 4194304);          // 4 MB

  wt_kernel<<<dim3(512, 3), 256, 0, stream>>>(Wq, Wk, Wv, Wt);
  qkv_kernel<<<NBATCH * SEQ / 16, 256, 0, stream>>>(x, Wt, Qb, Kb, Vt);
  attn_kernel<<<(NBATCH * SEQ / 16) / ATTN_WAVES, 32 * ATTN_WAVES, 0, stream>>>(Qb, Kb, Vt, out);
}
